// WindowAttention_29652454211861
// MI455X (gfx1250) — matrix operand order repaired and checked by reading, not compiled, not run on hardware
//
#include <hip/hip_runtime.h>

// ---- types ---------------------------------------------------------------
typedef __attribute__((ext_vector_type(16))) _Float16 v16h;
typedef __attribute__((ext_vector_type(8)))  _Float16 v8h;
typedef __attribute__((ext_vector_type(2)))  _Float16 v2h;
typedef __attribute__((ext_vector_type(8)))  float    v8f;
typedef __attribute__((ext_vector_type(4)))  float    v4f;

// ---- problem constants ---------------------------------------------------
#define NWIN   49      // tokens per window (7x7)
#define NP     64      // padded tokens per window
#define M2     128     // two windows stacked
#define CH     192     // channels
#define NH     6       // heads
#define HD     32      // head dim
#define J3     576     // 3*CH
#define NMASK  64      // nW

// LDS layout (bytes): two windows per block
#define OFF_Q   0                     // 128x192 f16  q (scaled+biased)
#define OFF_KO  49152                 // 128x192 f16  k; reused as O
#define OFF_VT  98304                 // 192x128 f16  v transposed
#define OFF_XS  147456                // union: X 128x192 f16 | S 6x64x64 f32 (one window)
#define LDS_BYTES (147456 + 98304)    // 245760 (240KB < 320KB/WGP)

__device__ __forceinline__ v8f wmma_f16(v16h a, v16h b, v8f c) {
  return __builtin_amdgcn_wmma_f32_16x16x32_f16(false, a, false, b,
                                                (short)0, c, false, false);
}

// A fragment (16x32 f16, row-major src, row stride rs elements)
__device__ __forceinline__ v16h load_a_frag(const _Float16* base, int rs, int lane) {
  const int r  = lane & 15;
  const int kb = (lane >> 4) * 8;          // 0 or 8
  const _Float16* p = base + r * rs + kb;
  v8h lo = *(const v8h*)(p);               // K = kb .. kb+7
  v8h hi = *(const v8h*)(p + 16);          // K = kb+16 .. kb+23
  v16h out;
#pragma unroll
  for (int i = 0; i < 8; ++i) { out[i] = lo[i]; out[8 + i] = hi[i]; }
  return out;
}

// A fragment gathered from f32 source (softmax probabilities)
__device__ __forceinline__ v16h load_a_frag_f32(const float* base, int rs, int lane) {
  const int r  = lane & 15;
  const int kb = (lane >> 4) * 8;
  const float* p = base + r * rs + kb;
  v4f a0 = *(const v4f*)(p);
  v4f a1 = *(const v4f*)(p + 4);
  v4f a2 = *(const v4f*)(p + 16);
  v4f a3 = *(const v4f*)(p + 20);
  v16h out;
#pragma unroll
  for (int i = 0; i < 4; ++i) {
    out[i]      = (_Float16)a0[i];
    out[4 + i]  = (_Float16)a1[i];
    out[8 + i]  = (_Float16)a2[i];
    out[12 + i] = (_Float16)a3[i];
  }
  return out;
}

// B fragment (32x16): src is B^T row-major [n][k], row stride rs
__device__ __forceinline__ v16h load_b_frag(const _Float16* baseT, int rs, int lane) {
  const int n  = lane & 15;
  const int kb = (lane >> 4) * 8;          // 0 or 8
  const _Float16* p = baseT + n * rs + kb;
  v8h lo = *(const v8h*)(p);
  v8h hi = *(const v8h*)(p + 16);
  v16h out;
#pragma unroll
  for (int i = 0; i < 8; ++i) { out[i] = lo[i]; out[8 + i] = hi[i]; }
  return out;
}

// ---- prep: f32->f16 weights + fused rel-pos bias table gather ------------
__global__ void winattn_prep(const float* __restrict__ qkv_w,
                             const float* __restrict__ proj_w,
                             const float* __restrict__ rel_table,
                             const int*   __restrict__ rel_index,
                             _Float16* __restrict__ qkvw_h,
                             _Float16* __restrict__ projw_h,
                             float*    __restrict__ biasW) {
  int i = blockIdx.x * blockDim.x + threadIdx.x;
  if (i < J3 * CH)  qkvw_h[i] = (_Float16)qkv_w[i];
  if (i < CH * CH)  projw_h[i] = (_Float16)proj_w[i];
  if (i < NH * NWIN * NWIN) {
    int h = i / (NWIN * NWIN);
    int nm = i % (NWIN * NWIN);
    biasW[i] = rel_table[rel_index[nm] * NH + h];
  }
}

// ---- main: one workgroup per TWO windows ---------------------------------
__global__ __launch_bounds__(256, 1)
void winattn_fwd(const float* __restrict__ x,
                 const float* __restrict__ mask,
                 const float* __restrict__ qkv_b,
                 const float* __restrict__ proj_b,
                 const _Float16* __restrict__ qkvw_h,
                 const _Float16* __restrict__ projw_h,
                 const float* __restrict__ biasW,
                 float* __restrict__ out) {
  extern __shared__ char smem[];
  _Float16* Qs  = (_Float16*)(smem + OFF_Q);
  _Float16* Ks  = (_Float16*)(smem + OFF_KO);
  _Float16* VTs = (_Float16*)(smem + OFF_VT);
  _Float16* Xs  = (_Float16*)(smem + OFF_XS);
  float*    Sf  = (float*)   (smem + OFF_XS);   // alias: live after X dies

  const int b0   = blockIdx.x * 2;              // first of two windows
  const int tid  = threadIdx.x;
  const int lane = tid & 31;
  const int wave = tid >> 5;

  // ---- phase 0: two windows of x -> f16 LDS, zero-pad n=49..63 ----
  // x is streamed once: non-temporal loads keep weights/bias/mask L2-hot.
  for (int i = tid; i < M2 * (CH / 4); i += 256) {
    int row = i / (CH / 4);                     // 0..127
    int cq  = (i % (CH / 4)) * 4;
    v4f v = {};
    int wi = row >> 6, n = row & 63;
    if (n < NWIN)
      v = __builtin_nontemporal_load(
            (const v4f*)(x + ((size_t)(b0 + wi) * NWIN + n) * CH + cq));
    v2h p0; p0[0] = (_Float16)v[0]; p0[1] = (_Float16)v[1];
    v2h p1; p1[0] = (_Float16)v[2]; p1[1] = (_Float16)v[3];
    *(v2h*)(Xs + row * CH + cq)     = p0;
    *(v2h*)(Xs + row * CH + cq + 2) = p1;
  }
  __syncthreads();

  // ---- phase 1: qkv = X @ qkv_w^T + b ; wave = fixed m-tile, stream j ----
  {
    const int mt = wave;                        // 8 waves <-> 8 m-tiles
    v16h a[6];
#pragma unroll
    for (int kc = 0; kc < 6; ++kc)
      a[kc] = load_a_frag(Xs + mt * 16 * CH + kc * 32, CH, lane);

    const float qscale = 0.17677669529663687f;  // 1/sqrt(32)
    const int jcol = lane & 15;
    const int mb   = mt * 16 + ((lane >> 4) * 8);
    for (int jt = 0; jt < 36; ++jt) {
      v8f acc0 = {}, acc1 = {};                 // two independent WMMA chains
#pragma unroll
      for (int kc = 0; kc < 6; kc += 2) {
        v16h bf0 = load_b_frag(qkvw_h + (jt * 16) * CH + kc * 32, CH, lane);
        v16h bf1 = load_b_frag(qkvw_h + (jt * 16) * CH + (kc + 1) * 32, CH, lane);
        acc0 = wmma_f16(a[kc],     bf0, acc0);
        acc1 = wmma_f16(a[kc + 1], bf1, acc1);
      }
      v8f acc;
#pragma unroll
      for (int r = 0; r < 8; ++r) acc[r] = acc0[r] + acc1[r];

      int j = jt * 16 + jcol;
      float bb = qkv_b[j];
      if (jt < 12) {                 // q
#pragma unroll
        for (int r = 0; r < 8; ++r)
          Qs[(mb + r) * CH + j] = (_Float16)((acc[r] + bb) * qscale);
      } else if (jt < 24) {          // k
        int jj = j - CH;
#pragma unroll
        for (int r = 0; r < 8; ++r)
          Ks[(mb + r) * CH + jj] = (_Float16)(acc[r] + bb);
      } else {                       // v transposed [d][m], stride 128
        int jj = j - 2 * CH;
#pragma unroll
        for (int r = 0; r < 8; ++r)
          VTs[jj * M2 + (mb + r)] = (_Float16)(acc[r] + bb);
      }
    }
  }
  __syncthreads();

  // ---- attention core, one window at a time (shared f32 S buffer) ----
  const int it = wave & 3;                      // n-tile of this wave
  const int hg = wave >> 2;                     // head group: heads hg*3..hg*3+2
  for (int wi = 0; wi < 2; ++wi) {
    const float* maskb = mask + (size_t)((b0 + wi) & (NMASK - 1)) * (NWIN * NWIN);

    // -- phase 2: S[h] = q k^T + bias + mask --
    {
      v16h aq[3];
#pragma unroll
      for (int hh = 0; hh < 3; ++hh)
        aq[hh] = load_a_frag(Qs + (wi * NP + it * 16) * CH + (hg * 3 + hh) * HD,
                             CH, lane);
      const int nb = it * 16 + ((lane >> 4) * 8);
#pragma unroll
      for (int hh = 0; hh < 3; ++hh) {
        int h = hg * 3 + hh;
#pragma unroll
        for (int jt = 0; jt < 4; ++jt) {
          v16h bf = load_b_frag(Ks + (wi * NP + jt * 16) * CH + h * HD, CH, lane);
          v8f acc = {};
          acc = wmma_f16(aq[hh], bf, acc);
          int mk = jt * 16 + (lane & 15);
          float* Sh = Sf + h * (NP * NP);
#pragma unroll
          for (int r = 0; r < 8; ++r) {
            int nq  = nb + r;
            float v = acc[r];
            if (mk >= NWIN) v = -1e30f;                  // padded keys
            else if (nq < NWIN)
              v += biasW[h * (NWIN * NWIN) + nq * NWIN + mk]
                 + maskb[nq * NWIN + mk];
            Sh[nq * NP + mk] = v;
          }
        }
      }
    }
    __syncthreads();

    // -- phase 3: softmax rows (in place, f32, float4-vectorized) --
    for (int row = tid; row < NH * NP; row += 256) {
      int h = row >> 6, n = row & 63;
      v4f* p = (v4f*)(Sf + h * (NP * NP) + n * NP);
      v4f vmx = {-1e30f, -1e30f, -1e30f, -1e30f};
#pragma unroll
      for (int m = 0; m < 16; ++m) {
        v4f t = p[m];
#pragma unroll
        for (int c = 0; c < 4; ++c) vmx[c] = fmaxf(vmx[c], t[c]);
      }
      float mx = fmaxf(fmaxf(vmx[0], vmx[1]), fmaxf(vmx[2], vmx[3]));
      v4f vsum = {};
#pragma unroll
      for (int m = 0; m < 16; ++m) {
        v4f t = p[m], e;
#pragma unroll
        for (int c = 0; c < 4; ++c) e[c] = __expf(t[c] - mx);
        p[m] = e;
#pragma unroll
        for (int c = 0; c < 4; ++c) vsum[c] += e[c];
      }
      float inv = 1.f / (vsum[0] + vsum[1] + vsum[2] + vsum[3]);
#pragma unroll
      for (int m = 0; m < 16; ++m) {
        v4f t = p[m];
#pragma unroll
        for (int c = 0; c < 4; ++c) t[c] *= inv;
        p[m] = t;
      }
    }
    __syncthreads();

    // -- phase 4: O = P @ V  (A from f32 probs, B from V^T) --
    {
      _Float16* Os = Ks;                        // reuse k buffer as O
      v16h pa[3][2];
#pragma unroll
      for (int hh = 0; hh < 3; ++hh)
#pragma unroll
        for (int kc = 0; kc < 2; ++kc)
          pa[hh][kc] = load_a_frag_f32(
              Sf + (hg * 3 + hh) * (NP * NP) + it * 16 * NP + kc * 32, NP, lane);

      const int nb = it * 16 + ((lane >> 4) * 8);
#pragma unroll
      for (int hh = 0; hh < 3; ++hh) {
        int h = hg * 3 + hh;
#pragma unroll
        for (int dt = 0; dt < 2; ++dt) {
          v16h bf0 = load_b_frag(VTs + (h * HD + dt * 16) * M2 + wi * NP, M2, lane);
          v16h bf1 = load_b_frag(VTs + (h * HD + dt * 16) * M2 + wi * NP + 32, M2, lane);
          v8f acc0 = {}, acc1 = {};             // independent chains
          acc0 = wmma_f16(pa[hh][0], bf0, acc0);
          acc1 = wmma_f16(pa[hh][1], bf1, acc1);
          int cc = h * HD + dt * 16 + (lane & 15);
#pragma unroll
          for (int r = 0; r < 8; ++r)
            Os[(wi * NP + nb + r) * CH + cc] = (_Float16)(acc0[r] + acc1[r]);
        }
      }
    }
    __syncthreads();
  }

  // ---- phase 5: out = O @ proj_w^T + proj_b ; wave = fixed m-tile ----
  {
    const _Float16* Os = Ks;
    const int mt = wave;                        // 8 m-tiles over 128 rows
    v16h ao[6];
#pragma unroll
    for (int kc = 0; kc < 6; ++kc)
      ao[kc] = load_a_frag(Os + mt * 16 * CH + kc * 32, CH, lane);

    const int mb = mt * 16 + ((lane >> 4) * 8);
    for (int jt = 0; jt < 12; ++jt) {
      v8f acc0 = {}, acc1 = {};                 // two independent WMMA chains
#pragma unroll
      for (int kc = 0; kc < 6; kc += 2) {
        v16h bf0 = load_b_frag(projw_h + (jt * 16) * CH + kc * 32, CH, lane);
        v16h bf1 = load_b_frag(projw_h + (jt * 16) * CH + (kc + 1) * 32, CH, lane);
        acc0 = wmma_f16(ao[kc],     bf0, acc0);
        acc1 = wmma_f16(ao[kc + 1], bf1, acc1);
      }
      int j = jt * 16 + (lane & 15);
      float pb = proj_b[j];
#pragma unroll
      for (int r = 0; r < 8; ++r) {
        int m  = mb + r;
        int wi = m >> 6, nq = m & 63;
        if (nq < NWIN)
          __builtin_nontemporal_store(
              acc0[r] + acc1[r] + pb,
              out + ((size_t)(b0 + wi) * NWIN + nq) * CH + j);
      }
    }
  }
}

// ---- host entry ----------------------------------------------------------
extern "C" void kernel_launch(void* const* d_in, const int* in_sizes, int n_in,
                              void* d_out, int out_size, void* d_ws, size_t ws_size,
                              hipStream_t stream) {
  (void)in_sizes; (void)n_in; (void)out_size; (void)ws_size;
  const float* x      = (const float*)d_in[0];
  const float* mask   = (const float*)d_in[1];
  const float* table  = (const float*)d_in[2];
  const float* qkv_w  = (const float*)d_in[3];
  const float* qkv_b  = (const float*)d_in[4];
  const float* proj_w = (const float*)d_in[5];
  const float* proj_b = (const float*)d_in[6];
  const int*   relidx = (const int*)d_in[7];
  float* out = (float*)d_out;

  char* ws = (char*)d_ws;
  _Float16* qkvw_h  = (_Float16*)(ws);                 // 576*192 f16 = 221184 B
  _Float16* projw_h = (_Float16*)(ws + 221184);        // 192*192 f16 =  73728 B
  float*    biasW   = (float*)   (ws + 294912);        // 6*49*49 f32 =  57624 B

  winattn_prep<<<432, 256, 0, stream>>>(qkv_w, proj_w, table, relidx,
                                        qkvw_h, projw_h, biasW);
  winattn_fwd<<<2048, 256, LDS_BYTES, stream>>>(x, mask, qkv_b, proj_b,
                                                qkvw_h, projw_h, biasW, out);
}
